// SSGRL_7773890805949
// MI455X (gfx1250) — compile-verified, matrix operand order checked by reading, not run on hardware
//
#include <hip/hip_runtime.h>
#include <hip/hip_bf16.h>

// ---------------------------------------------------------------------------
// SSGRL fused pipeline for MI455X (gfx1250, wave32, WMMA bf16 -> f32)
// ---------------------------------------------------------------------------

typedef __attribute__((ext_vector_type(16))) __bf16 v16bf;
typedef __attribute__((ext_vector_type(16))) short  v16s;
typedef __attribute__((ext_vector_type(8)))  short  v8s;
typedef __attribute__((ext_vector_type(8)))  float  v8f;

#define B_      4
#define C_      1024
#define HW_     196
#define N_      784      // B*H*W
#define L_      80
#define WD_     300
#define WDP_    320      // padded K for word-emb GEMM (mult of 32)
#define OUT_    2048
#define BL_     320      // B*L
#define GATE3_  3072     // stacked gate weight rows

__device__ __forceinline__ unsigned short f2bf(float f) {
    unsigned int u = __builtin_bit_cast(unsigned int, f);
    u += 0x7fffu + ((u >> 16) & 1u);      // round-to-nearest-even
    return (unsigned short)(u >> 16);
}
__device__ __forceinline__ float sigmoidf_(float x) { return 1.f / (1.f + __expf(-x)); }

// ------------------------- pack / fold kernels -----------------------------

__global__ void cvt_bf16_kernel(const float* __restrict__ in,
                                unsigned short* __restrict__ out, int n) {
    int i = blockIdx.x * blockDim.x + threadIdx.x;
    if (i < n) out[i] = f2bf(in[i]);
}

// pad K dimension with zeros (row-major rows x kout, source rows x kin)
__global__ void cvt_pad_kernel(const float* __restrict__ in,
                               unsigned short* __restrict__ out,
                               int rows, int kin, int kout) {
    int i = blockIdx.x * blockDim.x + threadIdx.x;
    if (i >= rows * kout) return;
    int r = i / kout, k = i - r * kout;
    out[i] = (k < kin) ? f2bf(in[r * kin + k]) : (unsigned short)0;
}

// wgb[n][k] = wX[n%C][k] + wX[n%C][C+k], X selected by n/C (w3,w4,w5), bf16
__global__ void pack_gates_kernel(const float* __restrict__ w3,
                                  const float* __restrict__ w4,
                                  const float* __restrict__ w5,
                                  unsigned short* __restrict__ out) {
    int i = blockIdx.x * blockDim.x + threadIdx.x;
    if (i >= GATE3_ * C_) return;
    int n = i >> 10, k = i & (C_ - 1);
    int sel = n >> 10, nn = n & (C_ - 1);
    const float* src = (sel == 0) ? w3 : (sel == 1) ? w4 : w5;
    float v = src[(size_t)nn * 2 * C_ + k] + src[(size_t)nn * 2 * C_ + C_ + k];
    out[i] = f2bf(v);
}

// v[k] = sum_j fca_w[j]*fc3_w[j][k];  v[1024] = sum_j fca_w[j]*fc3_b[j] + fca_b
__global__ void semantic_fold_kernel(const float* __restrict__ fc3_w,
                                     const float* __restrict__ fc3_b,
                                     const float* __restrict__ fca_w,
                                     const float* __restrict__ fca_b,
                                     float* __restrict__ v) {
    int k = blockIdx.x * blockDim.x + threadIdx.x;
    if (k < C_) {
        float s = 0.f;
        for (int j = 0; j < C_; ++j) s += fca_w[j] * fc3_w[(size_t)j * C_ + k];
        v[k] = s;
    } else if (k == C_) {
        float s = 0.f;
        for (int j = 0; j < C_; ++j) s += fca_w[j] * fc3_b[j];
        v[C_] = s + fca_b[0];
    }
}

// ------------------------------ WMMA GEMM ----------------------------------
// D(MxN,f32) = A(MxK,bf16 row-major) @ W(NxK,bf16 row-major)^T
// One wave per 16x16 output tile; K stepped by 32 per v_wmma_f32_16x16x32_bf16.
__global__ void gemm_bf16_wmma(const unsigned short* __restrict__ A,
                               const unsigned short* __restrict__ W,
                               float* __restrict__ D,
                               int M, int N, int K) {
    int wave   = blockIdx.x * (blockDim.x >> 5) + (threadIdx.x >> 5);
    int tilesN = N >> 4;
    int tm = wave / tilesN;
    int tn = wave - tm * tilesN;
    if (tm * 16 >= M) return;

    int lane = threadIdx.x & 31;
    int hi   = lane >> 4;          // K-half selector (ISA A/B layout)
    int mr   = lane & 15;          // row of A / row of W (== output col)

    const unsigned short* arow = A + (size_t)(tm * 16 + mr) * K;
    const unsigned short* wrow = W + (size_t)(tn * 16 + mr) * K;

    v8f acc = {0.f, 0.f, 0.f, 0.f, 0.f, 0.f, 0.f, 0.f};

    for (int k0 = 0; k0 < K; k0 += 32) {
        // elements 0..7 -> K = k0 + hi*8 + e ; elements 8..15 -> K = k0+16+hi*8+e
        v8s a0 = *(const v8s*)(arow + k0 + hi * 8);
        v8s a1 = *(const v8s*)(arow + k0 + 16 + hi * 8);
        v8s b0 = *(const v8s*)(wrow + k0 + hi * 8);
        v8s b1 = *(const v8s*)(wrow + k0 + 16 + hi * 8);
        v16s av, bv;
#pragma unroll
        for (int e = 0; e < 8; ++e) {
            av[e] = a0[e]; av[e + 8] = a1[e];
            bv[e] = b0[e]; bv[e + 8] = b1[e];
        }
        acc = __builtin_amdgcn_wmma_f32_16x16x32_bf16(
            false, __builtin_bit_cast(v16bf, av),
            false, __builtin_bit_cast(v16bf, bv),
            (short)0, acc, false, false);
    }

#pragma unroll
    for (int r = 0; r < 8; ++r) {
        int row = tm * 16 + r + hi * 8;   // D layout: VGPR r holds M=r / M=r+8
        D[(size_t)row * N + tn * 16 + mr] = acc[r];
    }
}

// ----------------------- semantic attention kernels ------------------------

// coeff[n*80+l] = sum_k tanh(fwh[n,k]*fwd[l,k]) * v[k] + v[1024]; one wave each
__global__ void coeff_kernel(const float* __restrict__ fwh,
                             const float* __restrict__ fwd,
                             const float* __restrict__ v,
                             float* __restrict__ coeff) {
    int wave = blockIdx.x * (blockDim.x >> 5) + (threadIdx.x >> 5);
    if (wave >= N_ * L_) return;
    int n = wave / L_, l = wave - n * L_;
    int lane = threadIdx.x & 31;
    const float* a = fwh + (size_t)n * C_;
    const float* b = fwd + (size_t)l * C_;
    float s = 0.f;
    for (int k = lane; k < C_; k += 32) s += tanhf(a[k] * b[k]) * v[k];
#pragma unroll
    for (int off = 16; off > 0; off >>= 1) s += __shfl_xor(s, off, 32);
    if (lane == 0) coeff[(size_t)n * L_ + l] = s + v[C_];
}

// softmax over contiguous 196-element rows (torch .view semantics preserved)
__global__ void softmax196_kernel(float* __restrict__ c) {
    __shared__ float red[256];
    float* p = c + (size_t)blockIdx.x * HW_;
    int t = threadIdx.x;
    float m = (t < HW_) ? p[t] : -1e30f;
    red[t] = m; __syncthreads();
    for (int s = 128; s > 0; s >>= 1) {
        if (t < s) red[t] = fmaxf(red[t], red[t + s]);
        __syncthreads();
    }
    float mx = red[0]; __syncthreads();
    float e = (t < HW_) ? __expf(p[t] - mx) : 0.f;
    red[t] = e; __syncthreads();
    for (int s = 128; s > 0; s >>= 1) {
        if (t < s) red[t] += red[t + s];
        __syncthreads();
    }
    float inv = 1.f / red[0];
    if (t < HW_) p[t] = e * inv;
}

// g[b,l,c] = sum_s cf[b*15680 + s*80 + l] * x[(b*196+s)*1024 + c]
// also seeds nodes (f32) and nodes_bf for the GGNN.
__global__ void attn_kernel(const float* __restrict__ cf,
                            const float* __restrict__ x,
                            float* __restrict__ g,
                            float* __restrict__ nodes,
                            unsigned short* __restrict__ nbf) {
    int bl = blockIdx.x;
    int b = bl / L_, l = bl - b * L_;
    int c0 = threadIdx.x * 4;
    const float* cb = cf + (size_t)b * HW_ * L_ + l;
    const float* xb = x + (size_t)b * HW_ * C_ + c0;
    float a0 = 0.f, a1 = 0.f, a2 = 0.f, a3 = 0.f;
    for (int s = 0; s < HW_; ++s) {
        float w = cb[(size_t)s * L_];
        const float* xr = xb + (size_t)s * C_;
        a0 += w * xr[0]; a1 += w * xr[1]; a2 += w * xr[2]; a3 += w * xr[3];
    }
    size_t o = (size_t)bl * C_ + c0;
    g[o] = a0; g[o + 1] = a1; g[o + 2] = a2; g[o + 3] = a3;
    nodes[o] = a0; nodes[o + 1] = a1; nodes[o + 2] = a2; nodes[o + 3] = a3;
    nbf[o] = f2bf(a0); nbf[o + 1] = f2bf(a1); nbf[o + 2] = f2bf(a2); nbf[o + 3] = f2bf(a3);
}

// ------------------------------ GGNN kernels -------------------------------

// a[b,i,c] = sum_j adj[i,j]*nodes[b,j,c]  (K=80: VALU, not WMMA-worthy); bf16 out
__global__ void adjprop_kernel(const float* __restrict__ adj,
                               const float* __restrict__ nodes,
                               unsigned short* __restrict__ abf) {
    __shared__ float arow[L_];
    int bi = blockIdx.x;
    int b = bi / L_, i = bi - b * L_;
    if (threadIdx.x < L_) arow[threadIdx.x] = adj[(size_t)i * L_ + threadIdx.x];
    __syncthreads();
    int c0 = threadIdx.x * 4;
    const float* nb = nodes + (size_t)b * L_ * C_ + c0;
    float a0 = 0.f, a1 = 0.f, a2 = 0.f, a3 = 0.f;
    for (int j = 0; j < L_; ++j) {
        float w = arow[j];
        const float* nr = nb + (size_t)j * C_;
        a0 += w * nr[0]; a1 += w * nr[1]; a2 += w * nr[2]; a3 += w * nr[3];
    }
    size_t o = (size_t)bi * C_ + c0;
    abf[o] = f2bf(a0); abf[o + 1] = f2bf(a1); abf[o + 2] = f2bf(a2); abf[o + 3] = f2bf(a3);
}

// z/r gates; rn_bf = bf16(r * nodes)
__global__ void gate1_kernel(const float* __restrict__ G,   // 320 x 3072
                             const float* __restrict__ U3,  // 320 x 1024
                             const float* __restrict__ b3,
                             const float* __restrict__ b4,
                             const float* __restrict__ u3b,
                             const float* __restrict__ nodes,
                             float* __restrict__ zbuf,
                             unsigned short* __restrict__ rnbf) {
    int i = blockIdx.x * blockDim.x + threadIdx.x;
    if (i >= BL_ * C_) return;
    int r = i >> 10, c = i & (C_ - 1);
    float u  = U3[i] + u3b[c];
    float z  = sigmoidf_(G[(size_t)r * GATE3_ + c] + b3[c] + u);
    float rr = sigmoidf_(G[(size_t)r * GATE3_ + C_ + c] + b4[c] + u);
    zbuf[i] = z;
    rnbf[i] = f2bf(rr * nodes[i]);
}

// hcand + state update; refresh nodes_bf
__global__ void gate2_kernel(const float* __restrict__ G,
                             const float* __restrict__ U5,
                             const float* __restrict__ b5,
                             const float* __restrict__ u5b,
                             const float* __restrict__ zbuf,
                             float* __restrict__ nodes,
                             unsigned short* __restrict__ nbf) {
    int i = blockIdx.x * blockDim.x + threadIdx.x;
    if (i >= BL_ * C_) return;
    int r = i >> 10, c = i & (C_ - 1);
    float hc = tanhf(G[(size_t)r * GATE3_ + 2 * C_ + c] + b5[c] + U5[i] + u5b[c]);
    float z  = zbuf[i];
    float nd = (1.f - z) * nodes[i] + z * hc;
    nodes[i] = nd;
    nbf[i]   = f2bf(nd);
}

// ------------------------------ output path --------------------------------

__global__ void featcat_kernel(const unsigned short* __restrict__ nbf,
                               const float* __restrict__ g,
                               unsigned short* __restrict__ feat) {
    int i = blockIdx.x * blockDim.x + threadIdx.x;
    if (i >= BL_ * 2 * C_) return;
    int r = i >> 11, c = i & (2 * C_ - 1);
    feat[i] = (c < C_) ? nbf[(size_t)r * C_ + c]
                       : f2bf(g[(size_t)r * C_ + (c - C_)]);
}

__global__ void out_epilogue_kernel(float* __restrict__ out,
                                    const float* __restrict__ bias) {
    int i = blockIdx.x * blockDim.x + threadIdx.x;
    if (i >= BL_ * OUT_) return;
    int c = i & (OUT_ - 1);
    out[i] = tanhf(out[i] + bias[c]);
}

// ------------------------------- host side ---------------------------------

static inline int cdiv(int a, int b) { return (a + b - 1) / b; }

extern "C" void kernel_launch(void* const* d_in, const int* in_sizes, int n_in,
                              void* d_out, int out_size, void* d_ws, size_t ws_size,
                              hipStream_t stream) {
    (void)in_sizes; (void)n_in; (void)out_size; (void)ws_size;

    const float* fmap   = (const float*)d_in[0];
    const float* wemb   = (const float*)d_in[1];
    const float* adj    = (const float*)d_in[2];
    const float* fc1_w  = (const float*)d_in[3];
    const float* fc2_w  = (const float*)d_in[4];
    const float* fc3_w  = (const float*)d_in[5];
    const float* fc3_b  = (const float*)d_in[6];
    const float* fca_w  = (const float*)d_in[7];
    const float* fca_b  = (const float*)d_in[8];
    const float* w3     = (const float*)d_in[9];
    const float* b3     = (const float*)d_in[10];
    const float* u3     = (const float*)d_in[11];
    const float* u3b    = (const float*)d_in[12];
    const float* w4     = (const float*)d_in[13];
    const float* b4     = (const float*)d_in[14];
    const float* w5     = (const float*)d_in[15];
    const float* b5     = (const float*)d_in[16];
    const float* u5     = (const float*)d_in[17];
    const float* u5b    = (const float*)d_in[18];
    const float* fco_w  = (const float*)d_in[19];
    const float* fco_b  = (const float*)d_in[20];
    float* dout = (float*)d_out;

    // ---- workspace carve (256B aligned) ----
    char* wp = (char*)d_ws;
    auto alloc = [&](size_t bytes) -> void* {
        void* p = (void*)wp;
        wp += (bytes + 255) & ~(size_t)255;
        return p;
    };
    unsigned short* xb    = (unsigned short*)alloc((size_t)N_ * C_ * 2);
    unsigned short* w1b   = (unsigned short*)alloc((size_t)C_ * C_ * 2);
    unsigned short* web   = (unsigned short*)alloc((size_t)L_ * WDP_ * 2);
    unsigned short* w2b   = (unsigned short*)alloc((size_t)C_ * WDP_ * 2);
    unsigned short* wgb   = (unsigned short*)alloc((size_t)GATE3_ * C_ * 2);
    unsigned short* u3w   = (unsigned short*)alloc((size_t)C_ * C_ * 2);
    unsigned short* u5w   = (unsigned short*)alloc((size_t)C_ * C_ * 2);
    unsigned short* fcob  = (unsigned short*)alloc((size_t)OUT_ * 2 * C_ * 2);
    unsigned short* abf   = (unsigned short*)alloc((size_t)BL_ * C_ * 2);
    unsigned short* nbf   = (unsigned short*)alloc((size_t)BL_ * C_ * 2);
    unsigned short* rnbf  = (unsigned short*)alloc((size_t)BL_ * C_ * 2);
    unsigned short* featb = (unsigned short*)alloc((size_t)BL_ * 2 * C_ * 2);
    float* fwh   = (float*)alloc((size_t)N_ * C_ * 4);
    float* fwd   = (float*)alloc((size_t)L_ * C_ * 4);
    float* vvec  = (float*)alloc((size_t)(C_ + 1) * 4);
    float* coeff = (float*)alloc((size_t)N_ * L_ * 4);
    float* g     = (float*)alloc((size_t)BL_ * C_ * 4);
    float* nodes = (float*)alloc((size_t)BL_ * C_ * 4);
    float* zbuf  = (float*)alloc((size_t)BL_ * C_ * 4);
    float* Gg    = (float*)alloc((size_t)BL_ * GATE3_ * 4);
    float* U3f   = (float*)alloc((size_t)BL_ * C_ * 4);
    float* U5f   = (float*)alloc((size_t)BL_ * C_ * 4);

    auto gemm = [&](const unsigned short* A, const unsigned short* W, float* D,
                    int M, int N, int K) {
        int waves = (M / 16) * (N / 16);
        gemm_bf16_wmma<<<cdiv(waves, 8), 256, 0, stream>>>(A, W, D, M, N, K);
    };

    // ---- pack & fold ----
    cvt_bf16_kernel<<<cdiv(N_ * C_, 256), 256, 0, stream>>>(fmap, xb, N_ * C_);
    cvt_bf16_kernel<<<cdiv(C_ * C_, 256), 256, 0, stream>>>(fc1_w, w1b, C_ * C_);
    cvt_pad_kernel<<<cdiv(L_ * WDP_, 256), 256, 0, stream>>>(wemb, web, L_, WD_, WDP_);
    cvt_pad_kernel<<<cdiv(C_ * WDP_, 256), 256, 0, stream>>>(fc2_w, w2b, C_, WD_, WDP_);
    pack_gates_kernel<<<cdiv(GATE3_ * C_, 256), 256, 0, stream>>>(w3, w4, w5, wgb);
    cvt_bf16_kernel<<<cdiv(C_ * C_, 256), 256, 0, stream>>>(u3, u3w, C_ * C_);
    cvt_bf16_kernel<<<cdiv(C_ * C_, 256), 256, 0, stream>>>(u5, u5w, C_ * C_);
    cvt_bf16_kernel<<<cdiv(OUT_ * 2 * C_, 256), 256, 0, stream>>>(fco_w, fcob, OUT_ * 2 * C_);
    semantic_fold_kernel<<<cdiv(C_ + 1, 256), 256, 0, stream>>>(fc3_w, fc3_b, fca_w, fca_b, vvec);

    // ---- semantic attention ----
    gemm(xb,  w1b, fwh, N_, C_, C_);      // f_wh = x @ fc1_w.T
    gemm(web, w2b, fwd, L_, C_, WDP_);    // f_wd = word_emb @ fc2_w.T
    coeff_kernel<<<cdiv(N_ * L_, 8), 256, 0, stream>>>(fwh, fwd, vvec, coeff);
    softmax196_kernel<<<B_ * L_, 256, 0, stream>>>(coeff);
    attn_kernel<<<BL_, 256, 0, stream>>>(coeff, fmap, g, nodes, nbf);

    // ---- GGNN (3 timesteps) ----
    for (int t = 0; t < 3; ++t) {
        adjprop_kernel<<<BL_, 256, 0, stream>>>(adj, nodes, abf);
        gemm(abf,  wgb, Gg,  BL_, GATE3_, C_);   // fused [w3s;w4s;w5s] gate GEMM
        gemm(nbf,  u3w, U3f, BL_, C_, C_);       // nodes @ u3.T
        gate1_kernel<<<cdiv(BL_ * C_, 256), 256, 0, stream>>>(Gg, U3f, b3, b4, u3b,
                                                              nodes, zbuf, rnbf);
        gemm(rnbf, u5w, U5f, BL_, C_, C_);       // (r*nodes) @ u5.T
        gate2_kernel<<<cdiv(BL_ * C_, 256), 256, 0, stream>>>(Gg, U5f, b5, u5b,
                                                              zbuf, nodes, nbf);
    }

    // ---- output head ----
    featcat_kernel<<<cdiv(BL_ * 2 * C_, 256), 256, 0, stream>>>(nbf, g, featb);
    gemm(featb, fcob, dout, BL_, OUT_, 2 * C_);  // feat @ fco_w.T
    out_epilogue_kernel<<<cdiv(BL_ * OUT_, 256), 256, 0, stream>>>(dout, fco_b);
}
